// BertSelfAttention_43808666419411
// MI455X (gfx1250) — matrix operand order repaired and checked by reading, not compiled, not run on hardware
//
#include <hip/hip_runtime.h>

// ---------------------------------------------------------------------------
// BERT self-attention, fused flash-style, bf16 WMMA path for MI455X (gfx1250)
//  - QKV projection: bf16 WMMA GEMM (f32 accumulate)
//  - Attention: flash-style, block-cooperative K/V staging in LDS via
//    async global->LDS copies (ASYNCcnt), double buffered.
// ---------------------------------------------------------------------------

#define B_   4
#define S_   2048
#define HID_ 768
#define NH_  12
#define HD_  64

typedef __bf16 bf16_t;
typedef __attribute__((ext_vector_type(16))) __bf16 v16bf;
typedef __attribute__((ext_vector_type(8)))  __bf16 v8bf;
typedef __attribute__((ext_vector_type(8)))  float  v8f;
typedef __attribute__((ext_vector_type(4)))  int    v4i;

#if defined(__has_builtin)
#  if __has_builtin(__builtin_amdgcn_global_load_async_to_lds_b128)
#    define HAS_ASYNC_LDS 1
#  endif
#endif
#ifndef HAS_ASYNC_LDS
#  define HAS_ASYNC_LDS 0
#endif

__device__ __forceinline__ v8f splat8(float x) {
    v8f r;
#pragma unroll
    for (int i = 0; i < 8; ++i) r[i] = x;
    return r;
}

__device__ __forceinline__ v16bf cat16(v8bf lo, v8bf hi) {
    v16bf r;
#pragma unroll
    for (int i = 0; i < 8; ++i) { r[i] = lo[i]; r[i + 8] = hi[i]; }
    return r;
}

// A-matrix fragment (16x32 bf16, row-major source with leading dim ld).
// ISA layout: lane<16 -> row m=lane, K={k0+0..7, k0+16..23};
//             lane>=16 -> row m=lane-16, K={k0+8..15, k0+24..31}.
__device__ __forceinline__ v16bf load_a_frag(const bf16_t* base, int ld,
                                             int row0, int k0, int lane) {
    const int m  = lane & 15;
    const int hi = lane >> 4;
    const bf16_t* p = base + (size_t)(row0 + m) * ld + k0 + hi * 8;
    v8bf lo = *(const v8bf*)(p);
    v8bf hh = *(const v8bf*)(p + 16);
    return cat16(lo, hh);
}

// B-matrix fragment (32x16 bf16). Source stored "N-major": row n holds the K
// values contiguously (leading dim ld). ISA layout: lane<16 -> col n=lane,
// K={k0+0..7, k0+16..23}; lane>=16 -> col n=lane-16, K={k0+8..15, k0+24..31}.
__device__ __forceinline__ v16bf load_b_frag(const bf16_t* base, int ld,
                                             int n0, int k0, int lane) {
    const int n  = lane & 15;
    const int hi = lane >> 4;
    const bf16_t* p = base + (size_t)(n0 + n) * ld + k0 + hi * 8;
    v8bf lo = *(const v8bf*)(p);
    v8bf hh = *(const v8bf*)(p + 16);
    return cat16(lo, hh);
}

__device__ __forceinline__ v8f wmma_bf16(v16bf a, v16bf b, v8f c) {
    // D = A(16x32) * B(32x16) + C, f32 accumulate
    return __builtin_amdgcn_wmma_f32_16x16x32_bf16(
        false, a, false, b, (short)0, c, false, false);
}

// Async 16B global -> LDS copy (ASYNCcnt), fallback: load + ds_store.
__device__ __forceinline__ void async_cp16(bf16_t* lds_dst, const bf16_t* gsrc) {
#if HAS_ASYNC_LDS
    __builtin_amdgcn_global_load_async_to_lds_b128(
        (__attribute__((address_space(1))) v4i*)gsrc,
        (__attribute__((address_space(3))) v4i*)lds_dst,
        0, 0);
#else
    *(v8bf*)lds_dst = *(const v8bf*)gsrc;
#endif
}

template <int N>
__device__ __forceinline__ void wait_async() {
#if defined(__has_builtin) && __has_builtin(__builtin_amdgcn_s_wait_asynccnt)
    __builtin_amdgcn_s_wait_asynccnt((unsigned short)N);
#else
    if constexpr (N == 0) asm volatile("s_wait_asynccnt 0" ::: "memory");
    else                  asm volatile("s_wait_asynccnt 4" ::: "memory");
#endif
}

// Row-wise (over the 16-lane n-dimension) butterfly reductions on C-layout.
__device__ __forceinline__ v8f rowmax16(v8f x) {
#pragma unroll
    for (int m = 1; m < 16; m <<= 1) {
#pragma unroll
        for (int i = 0; i < 8; ++i)
            x[i] = fmaxf(x[i], __shfl_xor(x[i], m, 16));
    }
    return x;
}
__device__ __forceinline__ v8f rowsum16(v8f x) {
#pragma unroll
    for (int m = 1; m < 16; m <<= 1) {
#pragma unroll
        for (int i = 0; i < 8; ++i)
            x[i] += __shfl_xor(x[i], m, 16);
    }
    return x;
}

// ---------------------------------------------------------------------------
// Kernel 0a: fp32 -> bf16 conversion of activations
// ---------------------------------------------------------------------------
__global__ void cvt_x_kernel(const float* __restrict__ X,
                             bf16_t* __restrict__ Xbf, int n) {
    int i = blockIdx.x * blockDim.x + threadIdx.x;
    if (i < n) Xbf[i] = (bf16_t)X[i];
}

// ---------------------------------------------------------------------------
// Kernel 0b: transpose + convert weights: Wt[mat][n][k] = W[mat][k][n]
// ---------------------------------------------------------------------------
__global__ void cvt_w_kernel(const float* __restrict__ Wq,
                             const float* __restrict__ Wk,
                             const float* __restrict__ Wv,
                             bf16_t* __restrict__ Wt) {
    int i = blockIdx.x * blockDim.x + threadIdx.x;
    if (i >= 3 * HID_ * HID_) return;
    int mat = i / (HID_ * HID_);
    int r   = i % (HID_ * HID_);
    int nn  = r / HID_;
    int kk  = r % HID_;
    const float* W = (mat == 0) ? Wq : (mat == 1) ? Wk : Wv;
    Wt[i] = (bf16_t)W[(size_t)kk * HID_ + nn];
}

// ---------------------------------------------------------------------------
// Kernel 1: fused QKV projection GEMM (bf16 WMMA, f32 accumulate)
//   Q,K -> [B, NH, S, HD] bf16 ;  V -> [B, NH, HD, S] bf16 (transposed)
// One wave per 16x16 output tile; 8 waves per block.
// ---------------------------------------------------------------------------
__global__ void qkv_gemm_kernel(const bf16_t* __restrict__ Xbf,
                                const bf16_t* __restrict__ Wt,
                                const float* __restrict__ bq,
                                const float* __restrict__ bk,
                                const float* __restrict__ bv,
                                bf16_t* __restrict__ Qh,
                                bf16_t* __restrict__ Kh,
                                bf16_t* __restrict__ Vt) {
    const int lane = threadIdx.x & 31;
    const int wid  = threadIdx.x >> 5;
    const int ntiles = HID_ / 16;            // 48
    const int mtiles = (B_ * S_) / 16;       // 512
    int g = blockIdx.x * 8 + wid;
    if (g >= 3 * mtiles * ntiles) return;    // wave-uniform

    int mat = g / (mtiles * ntiles);
    int rem = g % (mtiles * ntiles);
    int mt  = rem / ntiles;
    int nt  = rem % ntiles;

    const bf16_t* W = Wt + (size_t)mat * HID_ * HID_;
    const int row0 = mt * 16;
    const int n0   = nt * 16;

    v8f acc = splat8(0.0f);
#pragma unroll 4
    for (int k0 = 0; k0 < HID_; k0 += 32) {
        v16bf a = load_a_frag(Xbf, HID_, row0, k0, lane);
        v16bf b = load_b_frag(W,   HID_, n0,   k0, lane);
        acc = wmma_bf16(a, b, acc);
    }

    const float* bias = (mat == 0) ? bq : (mat == 1) ? bk : bv;
    const int n  = lane & 15;
    const int hi = lane >> 4;
    const float bval = bias[n0 + n];
    const int ncol = n0 + n;
    const int h = ncol / HD_;
    const int d = ncol % HD_;
    const int b_idx  = row0 / S_;            // 16 | S_, tile lies in one batch
    const int s_base = row0 % S_;

    if (mat < 2) {
        bf16_t* dst = (mat == 0) ? Qh : Kh;
#pragma unroll
        for (int i = 0; i < 8; ++i) {
            int s = s_base + i + 8 * hi;
            dst[((size_t)(b_idx * NH_ + h) * S_ + s) * HD_ + d] =
                (bf16_t)(acc[i] + bval);
        }
    } else {
#pragma unroll
        for (int i = 0; i < 8; ++i) {
            int s = s_base + i + 8 * hi;
            Vt[((size_t)(b_idx * NH_ + h) * HD_ + d) * S_ + s] =
                (bf16_t)(acc[i] + bval);
        }
    }
}

// ---------------------------------------------------------------------------
// Kernel 2: flash attention. Block = 128 threads = 4 waves; each wave owns a
// 16-query tile. K/V tiles (32 keys) are staged cooperatively into LDS with
// async global->LDS copies, double-buffered; online softmax per row.
// ---------------------------------------------------------------------------
__global__ void flash_attn_kernel(const bf16_t* __restrict__ Qh,
                                  const bf16_t* __restrict__ Kh,
                                  const bf16_t* __restrict__ Vt,
                                  const float* __restrict__ mask,
                                  float* __restrict__ out) {
    const int tid  = threadIdx.x;
    const int lane = tid & 31;
    const int wid  = tid >> 5;               // 0..3
    const int n    = lane & 15;
    const int hi   = lane >> 4;

    int blk  = blockIdx.x;
    int qblk = blk % (S_ / 64);
    int bh   = blk / (S_ / 64);
    int h    = bh % NH_;
    int b    = bh / NH_;
    int q0   = qblk * 64 + wid * 16;

    const bf16_t* Q = Qh + (size_t)(b * NH_ + h) * S_ * HD_;
    const bf16_t* K = Kh + (size_t)(b * NH_ + h) * S_ * HD_;
    const bf16_t* V = Vt + (size_t)(b * NH_ + h) * HD_ * S_;
    const float* mrow = mask + (size_t)b * S_;

    // LDS: double-buffered K tile [32 keys][64 d], V tile [64 d][32 keys],
    // plus per-wave P staging [16 q][32 keys].
    __shared__ __align__(32) bf16_t kbuf[2][32 * 64];   // 2 x 4KB
    __shared__ __align__(32) bf16_t vbuf[2][64 * 32];   // 2 x 4KB
    __shared__ __align__(32) bf16_t pst[4][16 * 32];    // 4KB
    bf16_t* myp = pst[wid];

    // Issue the 32-key tile at key0 into buffer `buf` (4 async 16B copies
    // per thread: 2 for K, 2 for V; 128 threads cover 4KB + 4KB).
    auto issue_tile = [&](int buf, int key0) {
#pragma unroll
        for (int rep = 0; rep < 2; ++rep) {
            int c  = tid + rep * 128;        // 0..255
            int kr = c >> 3, ko = (c & 7) * 8;   // K: 8 chunks per 64-elem row
            async_cp16(&kbuf[buf][c * 8], K + (size_t)(key0 + kr) * HD_ + ko);
            int vr = c >> 2, vo = (c & 3) * 8;   // V: 4 chunks per 32-elem row
            async_cp16(&vbuf[buf][c * 8], V + (size_t)vr * S_ + key0 + vo);
        }
    };

    // Q fragments held across the whole key loop (HD=64 -> two K=32 chunks)
    v16bf qa0 = load_a_frag(Q, HD_, q0,  0, lane);
    v16bf qa1 = load_a_frag(Q, HD_, q0, 32, lane);

    v8f acc0 = splat8(0.0f), acc1 = splat8(0.0f);
    v8f acc2 = splat8(0.0f), acc3 = splat8(0.0f);
    v8f m_run = splat8(-1e30f);
    v8f l_run = splat8(0.0f);
    const float scale = 0.125f;              // 1/sqrt(64)

    issue_tile(0, 0);

    for (int it = 0; it < S_ / 32; ++it) {
        const int key0 = it * 32;
        const int cur  = it & 1;

        if (key0 + 32 < S_) {
            issue_tile(cur ^ 1, key0 + 32);  // prefetch next tile
            wait_async<4>();                 // current tile's 4 issues landed
        } else {
            wait_async<0>();
        }
        __syncthreads();                     // tile visible to all waves

        const bf16_t* kb = kbuf[cur];
        const bf16_t* vb = vbuf[cur];

        // ---- scores: two 16-key tiles, each = QK^T over HD=64 ----
        v16bf kb00 = load_b_frag(kb, HD_,  0,  0, lane);
        v16bf kb01 = load_b_frag(kb, HD_,  0, 32, lane);
        v16bf kb10 = load_b_frag(kb, HD_, 16,  0, lane);
        v16bf kb11 = load_b_frag(kb, HD_, 16, 32, lane);

        v8f s0 = splat8(0.0f), s1 = splat8(0.0f);
        s0 = wmma_bf16(qa0, kb00, s0);
        s0 = wmma_bf16(qa1, kb01, s0);
        s1 = wmma_bf16(qa0, kb10, s1);
        s1 = wmma_bf16(qa1, kb11, s1);

        const float mv0 = mrow[key0 + n];
        const float mv1 = mrow[key0 + 16 + n];
#pragma unroll
        for (int i = 0; i < 8; ++i) {
            s0[i] = s0[i] * scale + mv0;
            s1[i] = s1[i] * scale + mv1;
        }

        // ---- online softmax (row-wise over keys) ----
        v8f tmax;
#pragma unroll
        for (int i = 0; i < 8; ++i) tmax[i] = fmaxf(s0[i], s1[i]);
        tmax = rowmax16(tmax);

        v8f mnew, alpha, p0, p1, psum;
#pragma unroll
        for (int i = 0; i < 8; ++i) {
            mnew[i]  = fmaxf(m_run[i], tmax[i]);
            alpha[i] = __expf(m_run[i] - mnew[i]);
            p0[i]    = __expf(s0[i] - mnew[i]);
            p1[i]    = __expf(s1[i] - mnew[i]);
            psum[i]  = p0[i] + p1[i];
        }
        v8f rs = rowsum16(psum);
#pragma unroll
        for (int i = 0; i < 8; ++i) {
            l_run[i] = l_run[i] * alpha[i] + rs[i];
            m_run[i] = mnew[i];
            acc0[i] *= alpha[i];
            acc1[i] *= alpha[i];
            acc2[i] *= alpha[i];
            acc3[i] *= alpha[i];
        }

        // ---- stage P (C-layout -> row-major bf16 in LDS) ----
#pragma unroll
        for (int i = 0; i < 8; ++i) {
            int r = i + 8 * hi;
            myp[r * 32 + n]      = (bf16_t)p0[i];
            myp[r * 32 + 16 + n] = (bf16_t)p1[i];
        }
        // in-wave cross-lane RAW through LDS: wait for DS writes to land
        asm volatile("s_wait_dscnt 0" ::: "memory");

        // ---- re-read P as A-fragment (16 queries x 32 keys) ----
        v16bf pa = load_a_frag(myp, 32, 0, 0, lane);

        // ---- context: P(16x32) @ V(32keys x 64dims), 4 d-tiles ----
        v16bf vb0 = load_b_frag(vb, 32,  0, 0, lane);
        v16bf vb1 = load_b_frag(vb, 32, 16, 0, lane);
        v16bf vb2 = load_b_frag(vb, 32, 32, 0, lane);
        v16bf vb3 = load_b_frag(vb, 32, 48, 0, lane);
        acc0 = wmma_bf16(pa, vb0, acc0);
        acc1 = wmma_bf16(pa, vb1, acc1);
        acc2 = wmma_bf16(pa, vb2, acc2);
        acc3 = wmma_bf16(pa, vb3, acc3);

        __syncthreads();                     // done reading cur before reuse
    }

    // ---- normalize and write out [B, S, HID] ----
    v8f inv;
#pragma unroll
    for (int i = 0; i < 8; ++i) inv[i] = 1.0f / l_run[i];

    float* obase = out + ((size_t)b * S_ + q0) * HID_ + h * HD_;
#pragma unroll
    for (int i = 0; i < 8; ++i) {
        int r = i + 8 * hi;
        float* orow = obase + (size_t)r * HID_;
        orow[n]      = acc0[i] * inv[i];
        orow[16 + n] = acc1[i] * inv[i];
        orow[32 + n] = acc2[i] * inv[i];
        orow[48 + n] = acc3[i] * inv[i];
    }
}

// ---------------------------------------------------------------------------
// Launch
// ---------------------------------------------------------------------------
extern "C" void kernel_launch(void* const* d_in, const int* in_sizes, int n_in,
                              void* d_out, int out_size, void* d_ws, size_t ws_size,
                              hipStream_t stream) {
    const float* X    = (const float*)d_in[0];  // [B,S,HID]
    const float* mask = (const float*)d_in[1];  // [B,1,1,S]
    const float* Wq   = (const float*)d_in[2];
    const float* bq   = (const float*)d_in[3];
    const float* Wk   = (const float*)d_in[4];
    const float* bk   = (const float*)d_in[5];
    const float* Wv   = (const float*)d_in[6];
    const float* bv   = (const float*)d_in[7];
    float* out = (float*)d_out;

    // workspace layout (bf16), 256B aligned slices
    char* ws = (char*)d_ws;
    auto align256 = [](size_t x) { return (x + 255) & ~(size_t)255; };
    size_t off = 0;
    const size_t xbf_b = (size_t)B_ * S_ * HID_ * sizeof(bf16_t);
    const size_t wt_b  = (size_t)3 * HID_ * HID_ * sizeof(bf16_t);
    const size_t qh_b  = (size_t)B_ * NH_ * S_ * HD_ * sizeof(bf16_t);

    bf16_t* Xbf = (bf16_t*)(ws + off); off = align256(off + xbf_b);
    bf16_t* Wt  = (bf16_t*)(ws + off); off = align256(off + wt_b);
    bf16_t* Qh  = (bf16_t*)(ws + off); off = align256(off + qh_b);
    bf16_t* Kh  = (bf16_t*)(ws + off); off = align256(off + qh_b);
    bf16_t* Vt  = (bf16_t*)(ws + off); off = align256(off + qh_b);
    (void)ws_size; (void)in_sizes; (void)n_in; (void)out_size;

    // 0a: convert activations
    {
        int n = B_ * S_ * HID_;
        cvt_x_kernel<<<(n + 255) / 256, 256, 0, stream>>>(X, Xbf, n);
    }
    // 0b: transpose + convert weights
    {
        int n = 3 * HID_ * HID_;
        cvt_w_kernel<<<(n + 255) / 256, 256, 0, stream>>>(Wq, Wk, Wv, Wt);
    }
    // 1: fused QKV GEMM
    {
        int waves  = 3 * ((B_ * S_) / 16) * (HID_ / 16);  // 73728
        int blocks = (waves + 7) / 8;                     // 9216
        qkv_gemm_kernel<<<blocks, 256, 0, stream>>>(Xbf, Wt, bq, bk, bv,
                                                    Qh, Kh, Vt);
    }
    // 2: flash attention
    {
        int blocks = B_ * NH_ * (S_ / 64);                // 1536
        flash_attn_kernel<<<blocks, 128, 0, stream>>>(Qh, Kh, Vt, mask, out);
    }
}
